// MeanPlannerActor_38362647888179
// MI455X (gfx1250) — matrix operand order repaired and checked by reading, not compiled, not run on hardware
//
#include <hip/hip_runtime.h>
#include <hip/hip_bf16.h>
#include <math.h>

// ---------------- problem constants ----------------
#define STATE_DIM 256
#define ACTION_DIM 2
#define HID 512
#define NH 10            // 2 * N_HEADS
#define NHEADS 5
#define POP 4096
#define ITERS 4
#define ELITE 819        // round(4096 * 0.2)
#define MT 32            // population rows per block in the MLP kernel
#define NPANEL 16        // 512 / 32 K-panels
#define PANEL_BYTES (HID * 32 * 2)   // 32 KB bf16 panel

typedef __bf16 bf16;
typedef __attribute__((ext_vector_type(16))) __bf16 v16bf;
typedef __attribute__((ext_vector_type(8)))  __bf16 v8bf;
typedef __attribute__((ext_vector_type(8)))  float  v8f;

// ---------------- one-time preprocessing ----------------

// fp32 -> bf16 conversion of W2, stored pre-swizzled per (head, K-panel) in the
// exact LDS fragment layout sW[n*32 + kk]:
//   dst[((h*16 + pi)*512 + n)*32 + kk] = W2[h, pi*32 + kk, n]
// so the MLP kernel's panel staging is a contiguous 32KB async memcpy.
__global__ __launch_bounds__(256) void k_conv_swizzle(const float* __restrict__ src,
                                                      bf16* __restrict__ dst, int n) {
    int i = blockIdx.x * 256 + threadIdx.x;
    if (i >= n) return;
    int kk = i & 31;
    int nn = (i >> 5) & (HID - 1);
    int pi = (i >> 14) & (NPANEL - 1);
    int h  = i >> 18;
    dst[i] = (bf16)src[((size_t)h * HID + pi * 32 + kk) * HID + nn];
}

// base1[h][j] = b1[h][j] + sum_i state[i] * W1[h][i][j]  (state part, fp32)
__global__ __launch_bounds__(256) void k_base1(const float* __restrict__ state,
                                               const float* __restrict__ W1,
                                               const float* __restrict__ b1,
                                               float* __restrict__ base1) {
    int h = blockIdx.x;
    const float* w = W1 + (size_t)h * (STATE_DIM + ACTION_DIM) * HID;
    for (int j = threadIdx.x; j < HID; j += 256) {
        float s = b1[h * HID + j];
        for (int i = 0; i < STATE_DIM; ++i)
            s = fmaf(state[i], w[(size_t)i * HID + j], s);
        base1[h * HID + j] = s;
    }
}

// carry = { mu[2], std[2], best_a[2], best_score }
__global__ void k_init_carry(float* __restrict__ carry) {
    if (threadIdx.x == 0) {
        carry[0] = 0.f; carry[1] = 0.f;
        carry[2] = 0.4f; carry[3] = 0.4f;
        carry[4] = 0.f; carry[5] = 0.f;
        carry[6] = -3.0e38f;
    }
}

// ---------------- per-iteration kernels ----------------

// a[p][d] = clip(mu[d] + std[d]*eps[iter][p][d], -1, 1)
__global__ __launch_bounds__(256) void k_sample(const float* __restrict__ noise,
                                                const float* __restrict__ carry,
                                                float* __restrict__ a, int iter) {
    int p = blockIdx.x * 256 + threadIdx.x;
    #pragma unroll
    for (int d = 0; d < ACTION_DIM; ++d) {
        float eps = noise[((size_t)iter * POP + p) * ACTION_DIM + d];
        float v = fmaf(carry[2 + d], eps, carry[d]);
        v = fminf(fmaxf(v, -1.0f), 1.0f);
        a[p * ACTION_DIM + d] = v;
    }
}

// Fused per-head MLP forward: layer1 (rank-2 update) -> layer2 (bf16 WMMA GEMM,
// weights streamed via double-buffered GLOBAL_LOAD_ASYNC_TO_LDS_B128) -> relu
// -> layer3 (GEMV).  grid = (POP/MT, NH), block = 256 (8 wave32s).
__global__ __launch_bounds__(256) void k_mlp(const float* __restrict__ W1,
                                             const float* __restrict__ base1,
                                             const float* __restrict__ A,
                                             const bf16*  __restrict__ W2p,
                                             const float* __restrict__ b2,
                                             const float* __restrict__ W3,
                                             const float* __restrict__ b3,
                                             float* __restrict__ qbuf) {
    __shared__ bf16  sH[MT * HID];     // 32x512 activations (h1, later reused for h2)
    __shared__ bf16  sW0[HID * 32];    // W2 K-panel buffer 0 (32 KB)
    __shared__ bf16  sW1[HID * 32];    // W2 K-panel buffer 1 (32 KB)
    __shared__ float sRed[MT * 8];

    const int head = blockIdx.y;
    const int p0   = blockIdx.x * MT;
    const int tid  = threadIdx.x;
    const int lane = tid & 31;
    const int wave = tid >> 5;         // 0..7, each wave owns a 64-column strip
    const int half = lane >> 4;
    const int l16  = lane & 15;

    // per-head pre-swizzled panel base (uniform -> SGPR)
    const bf16* W2h = W2p + (size_t)head * HID * HID;

    // ---- layer 1: h1 = relu(base1 + a0*W1[256,:] + a1*W1[257,:]) -> LDS bf16 ----
    {
        int r  = tid & (MT - 1);
        int jb = (tid >> 5) * 64;
        float a0 = A[(p0 + r) * ACTION_DIM + 0];
        float a1 = A[(p0 + r) * ACTION_DIM + 1];
        const float* w1a0 = W1 + ((size_t)head * (STATE_DIM + ACTION_DIM) + STATE_DIM) * HID;
        const float* w1a1 = w1a0 + HID;
        const float* bs   = base1 + head * HID;
        for (int j = jb; j < jb + 64; ++j) {
            float v = fmaf(a0, w1a0[j], fmaf(a1, w1a1[j], bs[j]));
            v = v > 0.f ? v : 0.f;
            sH[r * HID + j] = (bf16)v;
        }
    }

    // async panel copy: 32KB contiguous, 8 x b128 per thread, memory -> LDS
    auto issue_panel = [&](int pi, bf16* dstLds) {
        unsigned lbase = (unsigned)(unsigned long long)(void*)dstLds;  // LDS offset in addr[31:0]
        unsigned long long gbase =
            (unsigned long long)(const void*)(W2h + (size_t)pi * (HID * 32));
        unsigned o = (unsigned)tid * 16u;
        #pragma unroll
        for (int i = 0; i < 8; ++i) {
            unsigned lds_addr = lbase + o + (unsigned)i * 4096u;
            unsigned goff     = o + (unsigned)i * 4096u;
            asm volatile("global_load_async_to_lds_b128 %0, %1, %2"
                         :
                         : "v"(lds_addr), "v"(goff), "s"(gbase)
                         : "memory");
        }
    };

    issue_panel(0, sW0);               // prefetch panel 0 while layer1 finishes
    __syncthreads();                   // layer1 activations visible

    // ---- layer 2: 32x512 = (32x512) @ (512x512) via v_wmma_f32_16x16x32_bf16 ----
    v8f acc[2][4] = {};
    for (int pi = 0; pi < NPANEL; ++pi) {
        bf16* cur = (pi & 1) ? sW1 : sW0;
        bf16* nxt = (pi & 1) ? sW0 : sW1;
        asm volatile("s_wait_asynccnt 0x0" ::: "memory");  // panel pi landed in LDS
        __syncthreads();                                   // visible to all waves
        if (pi + 1 < NPANEL) issue_panel(pi + 1, nxt);     // overlap next copy with WMMA

        const int kb = pi * 32;
        // A fragments: contiguous-k order, same as B (lanes 0-15: K 0..15; 16-31: K 16..31)
        v16bf af[2];
        #pragma unroll
        for (int mt = 0; mt < 2; ++mt) {
            int r = mt * 16 + l16;
            af[mt] = *(const v16bf*)&sH[r * HID + kb + half * 16];
        }
        // B fragments: lane holds column n, 16 contiguous K values from cur[n*32 + half*16]
        #pragma unroll
        for (int nt = 0; nt < 4; ++nt) {
            int n = wave * 64 + nt * 16 + l16;
            v16bf bfB = *(const v16bf*)&cur[n * 32 + half * 16];
            #pragma unroll
            for (int mt = 0; mt < 2; ++mt) {
                acc[mt][nt] = __builtin_amdgcn_wmma_f32_16x16x32_bf16(
                    false, af[mt], false, bfB, (short)0, acc[mt][nt], false, false);
            }
        }
        __syncthreads();   // all reads of `cur` done before it is refilled at pi+2
    }

    // ---- bias + relu, spill h2 (bf16) back to sH ----
    #pragma unroll
    for (int mt = 0; mt < 2; ++mt) {
        #pragma unroll
        for (int nt = 0; nt < 4; ++nt) {
            int col = wave * 64 + nt * 16 + l16;
            float bb = b2[head * HID + col];
            #pragma unroll
            for (int r = 0; r < 8; ++r) {
                int row = mt * 16 + half * 8 + r;
                float v = acc[mt][nt][r] + bb;
                v = v > 0.f ? v : 0.f;
                sH[row * HID + col] = (bf16)v;
            }
        }
    }
    __syncthreads();

    // ---- layer 3: q[p] = h2[p,:] . W3[head,:] + b3 ----
    {
        int r   = tid >> 3;       // 0..31
        int seg = tid & 7;        // 8 segments of 64 k
        const float* w3 = W3 + head * HID;
        float s = 0.f;
        for (int k = seg * 64; k < seg * 64 + 64; ++k)
            s = fmaf((float)sH[r * HID + k], w3[k], s);
        sRed[r * 8 + seg] = s;
    }
    __syncthreads();
    if (tid < MT) {
        float s = 0.f;
        #pragma unroll
        for (int i = 0; i < 8; ++i) s += sRed[tid * 8 + i];
        qbuf[(size_t)head * POP + p0 + tid] = s + b3[head];
    }
}

// scores[p] = mean_h min(q1[h][p], q2[h][p])
__global__ __launch_bounds__(256) void k_scores(const float* __restrict__ qbuf,
                                                float* __restrict__ scores) {
    int p = blockIdx.x * 256 + threadIdx.x;
    float s = 0.f;
    #pragma unroll
    for (int h = 0; h < NHEADS; ++h)
        s += fminf(qbuf[(size_t)h * POP + p], qbuf[(size_t)(h + NHEADS) * POP + p]);
    scores[p] = s * (1.0f / NHEADS);
}

// Single-workgroup elite selection: bitonic sort (desc) of 4096 scores in LDS,
// deterministic tree-reduced elite mean/std (ddof=1), best tracking.
__global__ __launch_bounds__(1024) void k_select(const float* __restrict__ scores,
                                                 const float* __restrict__ a,
                                                 float* __restrict__ carry) {
    __shared__ float sv[POP];
    __shared__ int   si[POP];
    __shared__ float red[1024];
    int tid = threadIdx.x;

    for (int i = tid; i < POP; i += 1024) { sv[i] = scores[i]; si[i] = i; }
    __syncthreads();

    for (int k = 2; k <= POP; k <<= 1) {
        for (int j = k >> 1; j > 0; j >>= 1) {
            for (int i = tid; i < POP; i += 1024) {
                int ixj = i ^ j;
                if (ixj > i) {
                    bool descBlock = ((i & k) == 0);
                    float vi = sv[i], vx = sv[ixj];
                    bool doSwap = descBlock ? (vi < vx) : (vi > vx);
                    if (doSwap) {
                        sv[i] = vx; sv[ixj] = vi;
                        int t = si[i]; si[i] = si[ixj]; si[ixj] = t;
                    }
                }
            }
            __syncthreads();
        }
    }

    #pragma unroll
    for (int d = 0; d < ACTION_DIM; ++d) {
        float s = 0.f, sq = 0.f;
        for (int e = tid; e < ELITE; e += 1024) {
            float av = a[si[e] * ACTION_DIM + d];
            s += av; sq += av * av;
        }
        red[tid] = s; __syncthreads();
        for (int off = 512; off > 0; off >>= 1) {
            if (tid < off) red[tid] += red[tid + off];
            __syncthreads();
        }
        float tot = red[0]; __syncthreads();
        red[tid] = sq; __syncthreads();
        for (int off = 512; off > 0; off >>= 1) {
            if (tid < off) red[tid] += red[tid + off];
            __syncthreads();
        }
        float totsq = red[0]; __syncthreads();
        if (tid == 0) {
            float mu  = tot / (float)ELITE;
            float var = (totsq - (float)ELITE * mu * mu) / (float)(ELITE - 1);
            var = var > 0.f ? var : 0.f;
            float sd = sqrtf(var);
            sd = sd < 0.1f ? 0.1f : sd;
            carry[d] = mu;
            carry[2 + d] = sd;
        }
    }
    __syncthreads();
    if (tid == 0) {
        float top = sv[0];
        if (top > carry[6]) {
            carry[6] = top;
            carry[4] = a[si[0] * ACTION_DIM + 0];
            carry[5] = a[si[0] * ACTION_DIM + 1];
        }
    }
}

__global__ void k_out(const float* __restrict__ carry, float* __restrict__ out) {
    if (threadIdx.x < ACTION_DIM) out[threadIdx.x] = carry[4 + threadIdx.x];
}

// ---------------- host-side orchestration ----------------
extern "C" void kernel_launch(void* const* d_in, const int* in_sizes, int n_in,
                              void* d_out, int out_size, void* d_ws, size_t ws_size,
                              hipStream_t stream) {
    const float* state = (const float*)d_in[0];
    const float* noise = (const float*)d_in[1];
    const float* W1    = (const float*)d_in[2];
    const float* b1    = (const float*)d_in[3];
    const float* W2    = (const float*)d_in[4];
    const float* b2    = (const float*)d_in[5];
    const float* W3    = (const float*)d_in[6];
    const float* b3    = (const float*)d_in[7];

    char* ws = (char*)d_ws;
    size_t off = 0;
    bf16*  W2p    = (bf16*) (ws + off); off += (size_t)NH * HID * HID * sizeof(bf16);
    float* base1  = (float*)(ws + off); off += (size_t)NH * HID * sizeof(float);
    float* Abuf   = (float*)(ws + off); off += (size_t)POP * ACTION_DIM * sizeof(float);
    float* qbuf   = (float*)(ws + off); off += (size_t)NH * POP * sizeof(float);
    float* scores = (float*)(ws + off); off += (size_t)POP * sizeof(float);
    float* carry  = (float*)(ws + off); off += 8 * sizeof(float);
    (void)ws_size; (void)in_sizes; (void)n_in; (void)out_size;

    // one-time preprocessing
    {
        int n = NH * HID * HID;
        k_conv_swizzle<<<(n + 255) / 256, 256, 0, stream>>>(W2, W2p, n);
        k_base1<<<NH, 256, 0, stream>>>(state, W1, b1, base1);
        k_init_carry<<<1, 32, 0, stream>>>(carry);
    }

    // CEM loop (serial dependencies via stream order)
    for (int t = 0; t < ITERS; ++t) {
        k_sample<<<POP / 256, 256, 0, stream>>>(noise, carry, Abuf, t);
        dim3 grid(POP / MT, NH);
        k_mlp<<<grid, 256, 0, stream>>>(W1, base1, Abuf, W2p, b2, W3, b3, qbuf);
        k_scores<<<POP / 256, 256, 0, stream>>>(qbuf, scores);
        k_select<<<1, 1024, 0, stream>>>(scores, Abuf, carry);
    }

    k_out<<<1, 32, 0, stream>>>(carry, (float*)d_out);
}
